// BasicBlockA_24309514895539
// MI455X (gfx1250) — matrix operand order repaired and checked by reading, not compiled, not run on hardware
//
#include <hip/hip_runtime.h>

// ---------------------------------------------------------------------------
// Fused PixelCNN-ish block for MI455X (gfx1250, wave32, WMMA).
//   stage1: conv3x3 (3 -> 48ch, masked weights) + bias + ELU   -> h (LDS, bf16)
//   stage2: grouped conv3x3 (48 -> 48, 16 groups) mean-folded  -> out + res*x
// Both convs run as bf16 WMMA GEMMs with f32 accumulation.
// Branch-free fragment gathers via zero-padded LDS regions.
// ---------------------------------------------------------------------------

typedef __attribute__((ext_vector_type(16))) __bf16 v16bf;
typedef __attribute__((ext_vector_type(8)))  float  v8f;

#define HW 128

#define XS_ELEMS 1200            // 3 * 20 * 20 live x-tile elements
#define XS_TOTAL 1600            // + zero/dump pad (dump-slot pad-K gathers run past it)
#define HSTRIDE  384             // hs channel stride: 24 n-tiles * 16 px
#define HS_ELEMS (48 * HSTRIDE)  // 18432 (live + dup-pixel dump slots)
#define HS_TOTAL (HS_ELEMS + 288)// + zeroed pad for invalid-K gathers

union V16U { uint4 q[2]; v16bf v; };
union T16U { int4 q[4]; int s[16]; };
union F4U  { float4 v; float f[4]; };

static __device__ __forceinline__ __bf16 f2bf(float f) {
  union { float f; unsigned u; } v; v.f = f;
  unsigned r = (v.u + 0x7FFFu + ((v.u >> 16) & 1u)) >> 16;  // RNE-ish
  union { unsigned short s; __bf16 b; } o; o.s = (unsigned short)r;
  return o.b;
}

static __device__ __forceinline__ float softplus_f(float x) {
  return (x > 20.f) ? x : log1pf(expf(x));
}

// masked weight element: (w*mask0 + softplus(c)*mask1) * mask
static __device__ __forceinline__ float masked_w(const float* __restrict__ w,
                                                 const float* __restrict__ c,
                                                 int l, int i, int j, int kh, int kw) {
  int idx = (((l * 3 + i) * 3 + j) * 3 + kh) * 3 + kw;
  float v = (kh == 1 && kw == 1 && i == j) ? softplus_f(c[idx]) : w[idx];
  if (kh == 2) v = 0.f;
  else if (kh == 1) {
    if (j <= i) { if (kw >= 2) v = 0.f; }   // mask[i,:i+1,mid,mid+1:] = 0
    else        { if (kw >= 1) v = 0.f; }   // mask[i,i+1:,mid,mid:]  = 0
  }
  return v;
}

// A1: [48][32]  row m = l*3+i, col k = cin*9+kh*3+kw  (27 used, zero pad)
// A2: [16][448] row m = out-ch c (<3 valid), col k = l*27+cin*9+kh*3+kw, w2/16
__global__ void prep_weights_kernel(const float* __restrict__ w1, const float* __restrict__ c1,
                                    const float* __restrict__ w2, const float* __restrict__ c2,
                                    __bf16* __restrict__ A1, __bf16* __restrict__ A2) {
  int tid = blockIdx.x * blockDim.x + threadIdx.x;
  if (tid < 48 * 32) {
    int m = tid >> 5, k = tid & 31;
    float v = 0.f;
    if (k < 27) {
      int l = m / 3, i = m % 3;
      int cin = k / 9, kh = (k % 9) / 3, kw = k % 3;
      v = masked_w(w1, c1, l, i, cin, kh, kw);
    }
    A1[tid] = f2bf(v);
  }
  if (tid < 16 * 448) {
    int m = tid / 448, k = tid % 448;
    float v = 0.f;
    if (m < 3 && k < 432) {
      int l = k / 27, r = k % 27;
      int cin = r / 9, kh = (r % 9) / 3, kw = r % 3;
      v = masked_w(w2, c2, l, m, cin, kh, kw) * (1.0f / 16.0f);
    }
    A2[tid] = f2bf(v);
  }
}

__global__ __launch_bounds__(256) void fused_block_kernel(
    const float* __restrict__ x, const float* __restrict__ bias1,
    const float* __restrict__ resp,
    const __bf16* __restrict__ A1, const __bf16* __restrict__ A2,
    float* __restrict__ out) {

  __shared__ __attribute__((aligned(16))) __bf16 xs[XS_TOTAL];
  __shared__ __attribute__((aligned(16))) __bf16 hs[HS_TOTAL];
  __shared__ __attribute__((aligned(16))) int tab1[32];
  __shared__ __attribute__((aligned(16))) int tab2[448];

  const int tid = threadIdx.x;
  const int b   = blockIdx.z;
  const int y0  = blockIdx.y * 16;
  const int x0  = blockIdx.x * 16;
  const __bf16 bz = f2bf(0.f);

  // --- gather-offset tables (invalid K -> zeroed pad region, no branches) ---
  if (tid < 32) {
    int k = tid;
    tab1[k] = (k < 27) ? ((k / 9) * 400 + ((k % 9) / 3) * 20 + (k % 3)) : XS_ELEMS;
  }
  for (int i = tid; i < 448; i += 256) {
    int t = HS_ELEMS;
    if (i < 432) {
      int l = i / 27, r = i % 27;
      int cin = r / 9, kh = (r % 9) / 3, kw = r % 3;
      t = (l * 3 + cin) * HSTRIDE + kh * 18 + kw;
    }
    tab2[i] = t;
  }

  // --- zero hs pad region (never written by stage 1) ---
  for (int i = HS_ELEMS + tid; i < HS_TOTAL; i += 256) hs[i] = bz;

  // --- load x tile (20x20, halo 2) + zero tail/pad region ---
  for (int i = tid; i < XS_TOTAL; i += 256) {
    float v = 0.f;
    if (i < XS_ELEMS) {
      int c = i / 400, rr = (i % 400) / 20, cc = i % 20;
      int gy = y0 - 2 + rr, gx = x0 - 2 + cc;
      if (gy >= 0 && gy < HW && gx >= 0 && gx < HW)
        v = x[((b * 3 + c) * HW + gy) * HW + gx];
    }
    xs[i] = f2bf(v);
  }
  __syncthreads();

  const int lane = tid & 31;
  const int wave = tid >> 5;
  const int an   = lane & 15;        // M/N index within tile
  const int hi   = lane >> 4;        // lane-half selector
  const int akb  = hi * 8;           // A K-base for this half
  const int bkb  = hi * 16;          // B K-base for this half

  const float res = resp[0];
  const float rg  = res > 0.f ? res : 0.f;   // res * gate

  // ================= Stage 1: 48x32 GEMM over 18x18 halo region ==========
  // A fragments: two contiguous 16B halves per row -> b128 loads
  V16U a1f[3];
#pragma unroll
  for (int mt = 0; mt < 3; ++mt) {
    const __bf16* rp = A1 + (mt * 16 + an) * 32;
    a1f[mt].q[0] = *(const uint4*)(rp + akb);
    a1f[mt].q[1] = *(const uint4*)(rp + 16 + akb);
  }
  // bias preload: 8 consecutive floats per (mt, lane-half)
  F4U bia[3][2];
#pragma unroll
  for (int mt = 0; mt < 3; ++mt) {
    bia[mt][0].v = *(const float4*)(bias1 + mt * 16 + hi * 8);
    bia[mt][1].v = *(const float4*)(bias1 + mt * 16 + hi * 8 + 4);
  }

  for (int t = wave; t < 24; t += 8) {   // 24 n-tiles: uniform 3 per wave
    int p  = t * 16 + an;                // 0..383 (324+ are dump slots)
    int py = p / 18, px = p - py * 18;
    int base = py * 20 + px;
    T16U tt;
    const int4* tp = (const int4*)(tab1 + akb);
    tt.q[0] = tp[0]; tt.q[1] = tp[1]; tt.q[2] = tp[4]; tt.q[3] = tp[5];
    v16bf bf;
#pragma unroll
    for (int e = 0; e < 16; ++e) bf[e] = xs[tt.s[e] + base];
#pragma unroll
    for (int mt = 0; mt < 3; ++mt) {
      v8f acc = {};
      acc = __builtin_amdgcn_wmma_f32_16x16x32_bf16(false, a1f[mt].v, false, bf,
                                                    (short)0, acc, false, false);
      int hbase = (mt * 16 + hi * 8) * HSTRIDE + p;
#pragma unroll
      for (int r = 0; r < 8; ++r) {
        float v  = acc[r] + bia[mt][r >> 2].f[r & 3];
        float ev = __expf(v) - 1.0f;          // fast ELU (v_exp_f32)
        v = v > 0.f ? v : ev;
        hs[hbase + r * HSTRIDE] = f2bf(v);
      }
    }
  }
  __syncthreads();

  // ========== Stage 2: 16x448 GEMM (mean folded) over 16x16 tile ==========
  const __bf16* A2row = A2 + an * 448;
  for (int nt = wave * 2; nt < wave * 2 + 2; ++nt) {   // 16 n-tiles / 8 waves
    int p  = nt * 16 + an;
    int py = p >> 4, px = p & 15;
    int base = py * 18 + px;
    v8f acc = {};
    for (int kt = 0; kt < 14; ++kt) {
      V16U af;
      af.q[0] = *(const uint4*)(A2row + kt * 32 + akb);
      af.q[1] = *(const uint4*)(A2row + kt * 32 + 16 + akb);
      T16U tt;
      const int4* tp = (const int4*)(tab2 + kt * 32 + akb);
      tt.q[0] = tp[0]; tt.q[1] = tp[1]; tt.q[2] = tp[4]; tt.q[3] = tp[5];
      v16bf bf;
#pragma unroll
      for (int e = 0; e < 16; ++e) bf[e] = hs[tt.s[e] + base];
      acc = __builtin_amdgcn_wmma_f32_16x16x32_bf16(false, af.v, false, bf,
                                                    (short)0, acc, false, false);
    }
    if (hi == 0) {                         // rows 0..2 are the real channels
      int gidx = ((b * 3) * HW + (y0 + py)) * HW + (x0 + px);
#pragma unroll
      for (int r = 0; r < 3; ++r)
        out[gidx + r * HW * HW] = acc[r] + rg * x[gidx + r * HW * HW];
    }
  }
}

extern "C" void kernel_launch(void* const* d_in, const int* in_sizes, int n_in,
                              void* d_out, int out_size, void* d_ws, size_t ws_size,
                              hipStream_t stream) {
  const float* x    = (const float*)d_in[0];
  const float* w1   = (const float*)d_in[1];
  const float* c1   = (const float*)d_in[2];
  const float* b1   = (const float*)d_in[3];
  const float* w2   = (const float*)d_in[4];
  const float* c2   = (const float*)d_in[5];
  const float* resp = (const float*)d_in[6];

  __bf16* A1 = (__bf16*)d_ws;                       // 48*32  bf16 = 3 KB
  __bf16* A2 = (__bf16*)((char*)d_ws + 4096);       // 16*448 bf16 = 14 KB

  prep_weights_kernel<<<28, 256, 0, stream>>>(w1, c1, w2, c2, A1, A2);

  dim3 grid(HW / 16, HW / 16, 64);                  // 8 x 8 x 64 tiles
  fused_block_kernel<<<grid, 256, 0, stream>>>(x, b1, resp, A1, A2, (float*)d_out);
}
